// AscendQwen3Next_GatedDeltaNet_64630667870757
// MI455X (gfx1250) — matrix operand order repaired and checked by reading, not compiled, not run on hardware
//
#include <hip/hip_runtime.h>

// ---------------------------------------------------------------------------
// Qwen3-Next Gated DeltaNet prefill for MI455X (gfx1250, wave32, WMMA + TDM)
// L=4096 D=2048 HK=16 HV=32 DK=DV=128 RATIO=2
//  - GEMMs: bf16 operands (pre-converted / pre-transposed), f32 accum,
//    v_wmma_f32_16x16x32_bf16; tiles staged by Tensor Data Mover with
//    double-buffered LDS (TENSORcnt pipelining); 256x128 block tile,
//    64x64 wave tile (0.5 KB LDS read per WMMA).
//  - recurrence: chunked (C=64) gated delta rule, WMMA everywhere except a
//    tiny single-wave 16-row forward substitution; f32 state in LDS.
// ---------------------------------------------------------------------------

typedef __attribute__((ext_vector_type(16))) __bf16 v16bf;
typedef __attribute__((ext_vector_type(8)))  float  v8f;
typedef __attribute__((ext_vector_type(4)))  float  v4f;
typedef __attribute__((ext_vector_type(4)))  unsigned int v4u;
typedef __attribute__((ext_vector_type(2)))  unsigned int v2u;
typedef __attribute__((ext_vector_type(4)))  unsigned int u32x4;
typedef __attribute__((ext_vector_type(8)))  int i32x8;
typedef __attribute__((ext_vector_type(4)))  int i32x4;

#define Lq 4096
#define Dh 2048
#define HKC 16
#define HVC 32
#define DKC 128
#define DVC 128

__device__ __forceinline__ unsigned short f2bf(float f) {
    unsigned int u = __float_as_uint(f);
    u += 0x7FFFu + ((u >> 16) & 1u);           // round-to-nearest-even
    return (unsigned short)(u >> 16);
}
__device__ __forceinline__ float bf2f(unsigned short s) {
    return __uint_as_float(((unsigned int)s) << 16);
}
__device__ __forceinline__ v8f zero8() {
    v8f z = {0.f,0.f,0.f,0.f,0.f,0.f,0.f,0.f};
    return z;
}
__device__ __forceinline__ v8f wmma_bf16(v16bf a, v16bf b, v8f c) {
    return __builtin_amdgcn_wmma_f32_16x16x32_bf16(false, a, false, b, (short)0, c, false, false);
}

union Frag16 { v16bf bf; v4u q[2]; };

// A fragment (16x32 bf16, row-major LDS tile, row stride `ld` ushorts).
__device__ __forceinline__ v16bf frag_a(const unsigned short* p, int row0, int ld, int koff, int lane) {
    int hi = lane >> 4;
    const unsigned short* b = p + (size_t)(row0 + (lane & 15)) * ld + koff + hi * 8;
    Frag16 f;
    f.q[0] = *(const v4u*)(b);
    f.q[1] = *(const v4u*)(b + 16);
    return f.bf;
}
// B fragment (32x16 bf16) from column-major LDS tile ([N][K], row stride `ld`).
__device__ __forceinline__ v16bf frag_b(const unsigned short* p, int n0, int ld, int koff, int lane) {
    int hi = lane >> 4;
    const unsigned short* b = p + (size_t)(n0 + (lane & 15)) * ld + koff + hi * 8;
    Frag16 f;
    f.q[0] = *(const v4u*)(b);
    f.q[1] = *(const v4u*)(b + 16);
    return f.bf;
}

// ---------------------------------------------------------------------------
// TDM: DMA one 2D bf16 tile (tile1 rows x tile0 elems) into LDS.
// Pads 4 DWORDs after every 32 DWORDs stored => 64-elem bf16 rows land at a
// 144B stride (bank-conflict-free ds_load_b128, 16B aligned).
// Descriptor packing per CDNA5 ISA.
// ---------------------------------------------------------------------------
__device__ __forceinline__ void tdm_load_2d_bf16(
    unsigned lds_byte_off, const void* gaddr,
    unsigned dim0, unsigned dim1,          // tensor row length / row count (elems)
    unsigned tile0, unsigned tile1,        // tile row length / row count (elems)
    unsigned stride0)                      // tensor row stride (elems)
{
    unsigned long long ga = (unsigned long long)gaddr;
    u32x4 g0;
    g0[0] = 1u;                                              // count=1, user mode
    g0[1] = lds_byte_off;                                    // lds_addr
    g0[2] = (unsigned)(ga & 0xFFFFFFFFu);                    // global_addr lo
    g0[3] = (unsigned)((ga >> 32) & 0x01FFFFFFu) | (2u << 30); // ga hi | type=2
    i32x8 g1;
    g1[0] = (int)(0x10000u | (1u << 20) | (4u << 22) | (3u << 25)); // 2B data, pad en, 32DW interval, 4DW pad
    g1[1] = (int)((dim0 & 0xFFFFu) << 16);                   // tensor_dim0 lo16
    g1[2] = (int)(((dim0 >> 16) & 0xFFFFu) | ((dim1 & 0xFFFFu) << 16));
    g1[3] = (int)(((dim1 >> 16) & 0xFFFFu) | ((tile0 & 0xFFFFu) << 16));
    g1[4] = (int)(tile1 & 0xFFFFu);                          // tile_dim1, tile_dim2=0
    g1[5] = (int)stride0;                                    // tensor_dim0_stride lo32
    g1[6] = 0;                                               // stride0 hi16 | stride1 lo16
    g1[7] = 0;
    i32x4 z4 = {0, 0, 0, 0};
#if __clang_major__ >= 23
    i32x8 z8 = {0, 0, 0, 0, 0, 0, 0, 0};
    __builtin_amdgcn_tensor_load_to_lds(g0, g1, z4, z4, z8, 0);
#else
    __builtin_amdgcn_tensor_load_to_lds(g0, g1, z4, z4, 0);
#endif
}

__device__ __forceinline__ unsigned lds_off(const void* p) {
    return (unsigned)(unsigned long long)p;   // low 32 bits of flat LDS addr = LDS offset
}

// ---------------------------------------------------------------------------
// GEMM: C[MxN] f32 = A[MxK] bf16 @ Bt[NxK] bf16 (B pre-transposed).
// 256x128 block tile, BK=64, 256 threads (8 waves, 4x2), 64x64 wave tile,
// TDM double buffering.  M % 256 == 0, N % 128 == 0, K % 64 == 0.
// ---------------------------------------------------------------------------
#define GLD 72   // ushorts per LDS tile row (144B: 64 bf16 + 8 pad)

__global__ __launch_bounds__(256) void gemm_bf16_tdm_kernel(
    const unsigned short* __restrict__ A, const unsigned short* __restrict__ Bt,
    float* __restrict__ C, int M, int N, int K)
{
    __shared__ alignas(16) unsigned short As[2][256 * GLD];  // 73.7 KB
    __shared__ alignas(16) unsigned short Bs[2][128 * GLD];  // 36.9 KB
    const int tid = threadIdx.x, lane = tid & 31, wave = tid >> 5;
    const int wm = wave >> 1, wn = wave & 1;          // 4x2 wave grid
    const int m0 = blockIdx.y * 256, n0 = blockIdx.x * 128;
    const int nk = K >> 6;

    v8f acc[4][4];
    for (int i = 0; i < 4; ++i) for (int j = 0; j < 4; ++j) acc[i][j] = zero8();

    if (wave == 0) {   // prologue: tile 0 (TDM issues once per wave, EXEC ignored)
        tdm_load_2d_bf16(lds_off(&As[0][0]), A + (size_t)m0 * K, K, M, 64, 256, K);
        tdm_load_2d_bf16(lds_off(&Bs[0][0]), Bt + (size_t)n0 * K, K, N, 64, 128, K);
    }

    for (int kk = 0; kk < nk; ++kk) {
        const int cur = kk & 1, nxt = cur ^ 1;
        __syncthreads();                       // everyone done reading buf `nxt`
        if (wave == 0) {
            if (kk + 1 < nk) {
                tdm_load_2d_bf16(lds_off(&As[nxt][0]), A + (size_t)m0 * K + (size_t)(kk + 1) * 64,
                                 K, M, 64, 256, K);
                tdm_load_2d_bf16(lds_off(&Bs[nxt][0]), Bt + (size_t)n0 * K + (size_t)(kk + 1) * 64,
                                 K, N, 64, 128, K);
                __builtin_amdgcn_s_wait_tensorcnt(2);   // in-order: tile kk complete
            } else {
                __builtin_amdgcn_s_wait_tensorcnt(0);
            }
        }
        __syncthreads();                       // tile kk visible to all waves
        for (int ks = 0; ks < 2; ++ks) {
            v16bf bfr[4];
            for (int j = 0; j < 4; ++j) bfr[j] = frag_b(Bs[cur], wn * 64 + j * 16, GLD, ks * 32, lane);
            for (int i = 0; i < 4; ++i) {
                v16bf af = frag_a(As[cur], wm * 64 + i * 16, GLD, ks * 32, lane);
                for (int j = 0; j < 4; ++j)
                    acc[i][j] = wmma_bf16(af, bfr[j], acc[i][j]);
            }
        }
    }

    const int hi = lane >> 4, ln = lane & 15;
    for (int i = 0; i < 4; ++i)
        for (int j = 0; j < 4; ++j) {
            int mb = m0 + wm * 64 + i * 16 + hi * 8;
            int nb = n0 + wn * 64 + j * 16 + ln;
            for (int r = 0; r < 8; ++r)
                C[(size_t)(mb + r) * N + nb] = acc[i][j][r];
        }
}

// ---------------------------------------------------------------------------
// One-time f32 -> bf16 row-major conversion (grid-stride, 4 elems/thread).
// ---------------------------------------------------------------------------
__global__ void convert_bf16_kernel(const float* __restrict__ src, unsigned short* __restrict__ dst,
                                    long long n4)
{
    long long i = (long long)blockIdx.x * blockDim.x + threadIdx.x;
    long long stride = (long long)gridDim.x * blockDim.x;
    for (; i < n4; i += stride) {
        v4f v = *(const v4f*)(src + i * 4);
        v2u w;
        w.x = f2bf(v.x) | ((unsigned)f2bf(v.y) << 16);
        w.y = f2bf(v.z) | ((unsigned)f2bf(v.w) << 16);
        *(v2u*)(dst + i * 4) = w;
    }
}

// ---------------------------------------------------------------------------
// Weight transpose + convert: Wt[n][k] = bf16(W[k][n]).  32x32 LDS tiles.
// grid = (N/32, K/32), block = 256 (32x8).
// ---------------------------------------------------------------------------
__global__ __launch_bounds__(256) void transpose_bf16_kernel(
    const float* __restrict__ W, unsigned short* __restrict__ Wt, int K, int N)
{
    __shared__ unsigned short tile[32][33];
    const int n0 = blockIdx.x * 32, k0 = blockIdx.y * 32;
    const int tx = threadIdx.x & 31, ty = threadIdx.x >> 5;
    for (int j = 0; j < 4; ++j) {
        int kk = k0 + ty + j * 8;
        tile[tx][ty + j * 8] = f2bf(W[(size_t)kk * N + n0 + tx]);
    }
    __syncthreads();
    for (int j = 0; j < 4; ++j) {
        int n = n0 + ty + j * 8;
        Wt[(size_t)n * K + k0 + tx] = tile[ty + j * 8][tx];
    }
}

// ---------------------------------------------------------------------------
// beta/g projection: small N (32+32), VALU. grid=L, block=64.
// ---------------------------------------------------------------------------
__global__ void proj_ba_kernel(const float* __restrict__ h, const float* __restrict__ Wb,
                               const float* __restrict__ Wa, const float* __restrict__ A_log,
                               const float* __restrict__ dt_bias,
                               float* __restrict__ beta, float* __restrict__ g)
{
    const int row = blockIdx.x, t = threadIdx.x;
    const float* W = (t < 32) ? Wb : Wa;
    const int col = t & 31;
    const float* hr = h + (size_t)row * Dh;
    float acc = 0.f;
    for (int k = 0; k < Dh; ++k) acc += hr[k] * W[(size_t)k * HVC + col];
    if (t < 32) {
        beta[(size_t)row * HVC + col] = 1.f / (1.f + __expf(-acc));
    } else {
        float x = acc + dt_bias[col];
        float sp = (x > 20.f) ? x : log1pf(__expf(x));
        g[(size_t)row * HVC + col] = -__expf(A_log[col]) * sp;   // log-decay (<=0)
    }
}

// ---------------------------------------------------------------------------
// L2-norm q and k per (token, k-head); fold DK^-0.5 into q. grid=L*HK, block=128.
// ---------------------------------------------------------------------------
__global__ void l2norm_qk_kernel(float* __restrict__ q, float* __restrict__ k)
{
    __shared__ float red[8];
    const int t = threadIdx.x;
    const size_t base = (size_t)blockIdx.x * DKC;
    float qv = q[base + t], kv = k[base + t];
    float sq = qv * qv, sk = kv * kv;
    for (int m = 16; m >= 1; m >>= 1) { sq += __shfl_xor(sq, m, 32); sk += __shfl_xor(sk, m, 32); }
    if ((t & 31) == 0) { red[t >> 5] = sq; red[4 + (t >> 5)] = sk; }
    __syncthreads();
    sq = red[0] + red[1] + red[2] + red[3];
    sk = red[4] + red[5] + red[6] + red[7];
    q[base + t] = qv * rsqrtf(sq + 1e-6f) * 0.08838834764831845f;  // * 128^-0.5
    k[base + t] = kv * rsqrtf(sk + 1e-6f);
}

// ---------------------------------------------------------------------------
// Chunked gated delta-rule recurrence. grid = (HV=32, 2 dv-halves), block 256.
// ---------------------------------------------------------------------------
#define LHD 136   // ld ushorts for [*][128] bf16 tiles (padded)
#define LHT 72    // ld ushorts for [*][64]  bf16 tiles (padded)
#define LF  68    // ld floats  for [*][64]  f32  tiles (padded)

__global__ __launch_bounds__(256) void gdn_recurrence_kernel(
    const float* __restrict__ q, const float* __restrict__ k, const float* __restrict__ v,
    const float* __restrict__ beta, const float* __restrict__ g, float* __restrict__ core)
{
    __shared__ alignas(16) float          Sf[128 * LF];   // S[dk][dv_local] f32
    __shared__ alignas(16) unsigned short Sbf[64 * LHD];  // S^T bf16 [dv][dk]  (B operand)
    __shared__ alignas(16) unsigned short Kc[64 * LHD];   // k chunk [t][dk]
    __shared__ alignas(16) unsigned short Qc[64 * LHD];   // q chunk [t][dk]
    __shared__ alignas(16) unsigned short KhT[128 * LHT]; // (giv*k)^T [dk][t]
    __shared__ alignas(16) unsigned short Ut[64 * LHT];   // U^T bf16 [dv][t]   (B operand)
    __shared__ alignas(16) unsigned short Mb[64 * LHT];   // M bf16 [t][i]
    __shared__ alignas(16) unsigned short Ab[64 * LHT];   // Attn bf16 [t][i]
    __shared__ alignas(16) float          Rf[64 * LF];    // residual / U f32 [t][dv]
    __shared__ float gcs[64], gam[64], bet[64], giv[64];
    __shared__ float GammaC;

    const int h = blockIdx.x;
    const int c0 = blockIdx.y * 64;
    const int kh = h >> 1;                 // GQA ratio 2
    const int tid = threadIdx.x, lane = tid & 31, wave = tid >> 5;
    const int hi = lane >> 4, ln = lane & 15;

    for (int i = tid; i < 128 * LF; i += 256) Sf[i] = 0.f;
    __syncthreads();

    for (int cc = 0; cc < Lq / 64; ++cc) {
        const int t0 = cc * 64;

        // --- phase A: per-token scalars -----------------------------------
        if (tid < 64) gcs[tid] = g[(size_t)(t0 + tid) * HVC + h];
        __syncthreads();
        if (tid == 0) {
            float cs = 0.f;
            for (int t = 0; t < 64; ++t) { cs += gcs[t]; gcs[t] = cs; gam[t] = __expf(cs); }
            GammaC = gam[63];
        }
        __syncthreads();
        if (tid < 64) {
            bet[tid] = beta[(size_t)(t0 + tid) * HVC + h];
            giv[tid] = __expf(gcs[63] - gcs[tid]);       // Gamma/gam_t  (<=1)
        }
        __syncthreads();

        // --- phase B: stage chunk operands in LDS (f32 -> bf16) -----------
        for (int idx = tid; idx < 64 * 128; idx += 256) {
            int t = idx >> 7, d = idx & 127;
            float kv = k[(size_t)(t0 + t) * (HKC * DKC) + kh * DKC + d];
            float qv = q[(size_t)(t0 + t) * (HKC * DKC) + kh * DKC + d];
            Kc[t * LHD + d]  = f2bf(kv);
            Qc[t * LHD + d]  = f2bf(qv);
            KhT[d * LHT + t] = f2bf(kv * giv[t]);
        }
        for (int idx = tid; idx < 64 * 128; idx += 256) {
            int dvi = idx >> 7, d = idx & 127;
            Sbf[dvi * LHD + d] = f2bf(Sf[d * LF + dvi]);  // transpose to [dv][dk]
        }
        for (int idx = tid; idx < 64 * LHT; idx += 256) Ut[idx] = 0;
        __syncthreads();

        // --- phase C: P=K@K^T -> M ; Attn=Q@K^T ; R0=K@S -> Rf ------------
        for (int fi = wave; fi < 16; fi += 8) {             // M (strictly lower, scaled)
            int fm = fi >> 2, fn = fi & 3;
            v8f acc = zero8();
            for (int kk = 0; kk < 4; ++kk)
                acc = wmma_bf16(frag_a(Kc, fm * 16, LHD, kk * 32, lane),
                                frag_b(Kc, fn * 16, LHD, kk * 32, lane), acc);
            for (int r = 0; r < 8; ++r) {
                int m = fm * 16 + hi * 8 + r, n = fn * 16 + ln;
                unsigned short o = 0;
                if (n < m) o = f2bf(bet[m] * acc[r] * __expf(gcs[m] - gcs[n]));
                Mb[m * LHT + n] = o;
            }
        }
        for (int fi = wave; fi < 16; fi += 8) {             // Attn (incl. diagonal)
            int fm = fi >> 2, fn = fi & 3;
            v8f acc = zero8();
            for (int kk = 0; kk < 4; ++kk)
                acc = wmma_bf16(frag_a(Qc, fm * 16, LHD, kk * 32, lane),
                                frag_b(Kc, fn * 16, LHD, kk * 32, lane), acc);
            for (int r = 0; r < 8; ++r) {
                int m = fm * 16 + hi * 8 + r, n = fn * 16 + ln;
                unsigned short o = 0;
                if (n <= m) o = f2bf(acc[r] * __expf(gcs[m] - gcs[n]));
                Ab[m * LHT + n] = o;
            }
        }
        for (int fi = wave; fi < 16; fi += 8) {             // Rf = beta*(V - gam*K@S)
            int fm = fi >> 2, fn = fi & 3;
            v8f acc = zero8();
            for (int kk = 0; kk < 4; ++kk)
                acc = wmma_bf16(frag_a(Kc, fm * 16, LHD, kk * 32, lane),
                                frag_b(Sbf, fn * 16, LHD, kk * 32, lane), acc);
            for (int r = 0; r < 8; ++r) {
                int m = fm * 16 + hi * 8 + r, n = fn * 16 + ln;
                float vv = v[(size_t)(t0 + m) * (HVC * DVC) + h * DVC + c0 + n];
                Rf[m * LF + n] = bet[m] * (vv - gam[m] * acc[r]);
            }
        }
        __syncthreads();

        // --- phase D: blocked forward substitution for U ------------------
        for (int b = 0; b < 4; ++b) {
            if (b > 0) {
                if (wave < 4) {   // cross-block correction via WMMA (later Ut blocks are 0)
                    int fn = wave;
                    v8f acc = zero8();
                    for (int kk = 0; kk < 2; ++kk)
                        acc = wmma_bf16(frag_a(Mb, b * 16, LHT, kk * 32, lane),
                                        frag_b(Ut, fn * 16, LHT, kk * 32, lane), acc);
                    for (int r = 0; r < 8; ++r) {
                        int m = b * 16 + hi * 8 + r, n = fn * 16 + ln;
                        Rf[m * LF + n] -= acc[r];
                    }
                }
                __syncthreads();
            }
            if (wave == 0) {      // 16 serial rows, single wave => LDS in-order
                for (int t = b * 16; t < b * 16 + 16; ++t) {
                    for (int half = 0; half < 2; ++half) {
                        int n = lane + half * 32;
                        float a2 = Rf[t * LF + n];
                        for (int i = b * 16; i < t; ++i)
                            a2 -= bf2f(Mb[t * LHT + i]) * Rf[i * LF + n];
                        Rf[t * LF + n] = a2;
                        Ut[n * LHT + t] = f2bf(a2);
                    }
                }
            }
            __syncthreads();
        }

        // --- phase E: outputs + state update ------------------------------
        for (int fi = wave; fi < 16; fi += 8) {  // O = gam*(Q@S_prev) + Attn@U
            int fm = fi >> 2, fn = fi & 3;
            v8f aI = zero8(), aO = zero8();
            for (int kk = 0; kk < 2; ++kk)
                aI = wmma_bf16(frag_a(Ab, fm * 16, LHT, kk * 32, lane),
                               frag_b(Ut, fn * 16, LHT, kk * 32, lane), aI);
            for (int kk = 0; kk < 4; ++kk)
                aO = wmma_bf16(frag_a(Qc, fm * 16, LHD, kk * 32, lane),
                               frag_b(Sbf, fn * 16, LHD, kk * 32, lane), aO);
            for (int r = 0; r < 8; ++r) {
                int m = fm * 16 + hi * 8 + r, n = fn * 16 + ln;
                core[(size_t)(t0 + m) * (HVC * DVC) + h * DVC + c0 + n] = gam[m] * aO[r] + aI[r];
            }
        }
        float Gc = GammaC;
        for (int fi = wave; fi < 32; fi += 8) {  // S = Gamma*S + KhT @ U
            int fm = fi >> 2, fn = fi & 3;       // fm: dk frag 0..7, fn: dv frag 0..3
            v8f acc = zero8();
            for (int kk = 0; kk < 2; ++kk)
                acc = wmma_bf16(frag_a(KhT, fm * 16, LHT, kk * 32, lane),
                                frag_b(Ut, fn * 16, LHT, kk * 32, lane), acc);
            for (int r = 0; r < 8; ++r) {
                int m = fm * 16 + hi * 8 + r, n = fn * 16 + ln;
                Sf[m * LF + n] = Gc * Sf[m * LF + n] + acc[r];
            }
        }
        __syncthreads();
    }
}

// ---------------------------------------------------------------------------
// Gated RMSNorm: x = core*silu(z); x *= rsqrt(mean(x^2)+eps)*w. Emits bf16.
// grid=L*HV, blk=128.
// ---------------------------------------------------------------------------
__global__ void gated_rmsnorm_kernel(const float* __restrict__ core, const float* __restrict__ z,
                                     const float* __restrict__ w, unsigned short* __restrict__ xn)
{
    __shared__ float red[4];
    const int t = threadIdx.x;
    const size_t base = (size_t)blockIdx.x * DVC;
    float zv = z[base + t];
    float x = core[base + t] * (zv / (1.f + __expf(-zv)));
    float s = x * x;
    for (int m = 16; m >= 1; m >>= 1) s += __shfl_xor(s, m, 32);
    if ((t & 31) == 0) red[t >> 5] = s;
    __syncthreads();
    s = red[0] + red[1] + red[2] + red[3];
    xn[base + t] = f2bf(x * rsqrtf(s * (1.f / 128.f) + 1e-6f) * w[t]);
}

// ---------------------------------------------------------------------------
extern "C" void kernel_launch(void* const* d_in, const int* in_sizes, int n_in,
                              void* d_out, int out_size, void* d_ws, size_t ws_size,
                              hipStream_t stream)
{
    (void)in_sizes; (void)n_in; (void)out_size; (void)ws_size;
    const float* h       = (const float*)d_in[0];
    const float* Wq      = (const float*)d_in[1];
    const float* Wk      = (const float*)d_in[2];
    const float* Wv      = (const float*)d_in[3];
    const float* Wz      = (const float*)d_in[4];
    const float* Wb      = (const float*)d_in[5];
    const float* Wa      = (const float*)d_in[6];
    const float* A_log   = (const float*)d_in[7];
    const float* dt_bias = (const float*)d_in[8];
    const float* norm_w  = (const float*)d_in[9];
    const float* W_out   = (const float*)d_in[10];
    float* out = (float*)d_out;

    // ---- workspace layout -------------------------------------------------
    float* ws    = (float*)d_ws;
    float* qb    = ws;                               // L*2048 f32
    float* kb    = qb    + (size_t)Lq * 2048;
    float* vb    = kb    + (size_t)Lq * 2048;        // L*4096 f32
    float* zb    = vb    + (size_t)Lq * 4096;
    float* betab = zb    + (size_t)Lq * 4096;        // L*32
    float* gb    = betab + (size_t)Lq * 32;
    float* coreb = gb    + (size_t)Lq * 32;          // L*4096 f32
    unsigned short* hb  = (unsigned short*)(coreb + (size_t)Lq * 4096);  // L*2048 bf16
    unsigned short* xnb = hb  + (size_t)Lq * 2048;                       // L*4096 bf16
    unsigned short* WqT = xnb + (size_t)Lq * 4096;   // [2048][2048] bf16 (N-major)
    unsigned short* WkT = WqT + (size_t)2048 * 2048;
    unsigned short* WvT = WkT + (size_t)2048 * 2048; // [4096][2048]
    unsigned short* WzT = WvT + (size_t)4096 * 2048;
    unsigned short* WoT = WzT + (size_t)4096 * 2048; // [2048][4096]

    // ---- one-time bf16 conversion / weight transposition ------------------
    convert_bf16_kernel<<<1024, 256, 0, stream>>>(h, hb, (long long)Lq * 2048 / 4);
    transpose_bf16_kernel<<<dim3(2048 / 32, 2048 / 32), 256, 0, stream>>>(Wq, WqT, 2048, 2048);
    transpose_bf16_kernel<<<dim3(2048 / 32, 2048 / 32), 256, 0, stream>>>(Wk, WkT, 2048, 2048);
    transpose_bf16_kernel<<<dim3(4096 / 32, 2048 / 32), 256, 0, stream>>>(Wv, WvT, 2048, 4096);
    transpose_bf16_kernel<<<dim3(4096 / 32, 2048 / 32), 256, 0, stream>>>(Wz, WzT, 2048, 4096);
    transpose_bf16_kernel<<<dim3(2048 / 32, 4096 / 32), 256, 0, stream>>>(W_out, WoT, 4096, 2048);

    // ---- projections (TDM + WMMA GEMMs) -----------------------------------
    gemm_bf16_tdm_kernel<<<dim3(2048 / 128, Lq / 256), 256, 0, stream>>>(hb, WqT, qb, Lq, 2048, Dh);
    gemm_bf16_tdm_kernel<<<dim3(2048 / 128, Lq / 256), 256, 0, stream>>>(hb, WkT, kb, Lq, 2048, Dh);
    gemm_bf16_tdm_kernel<<<dim3(4096 / 128, Lq / 256), 256, 0, stream>>>(hb, WvT, vb, Lq, 4096, Dh);
    gemm_bf16_tdm_kernel<<<dim3(4096 / 128, Lq / 256), 256, 0, stream>>>(hb, WzT, zb, Lq, 4096, Dh);
    proj_ba_kernel<<<Lq, 64, 0, stream>>>(h, Wb, Wa, A_log, dt_bias, betab, gb);
    l2norm_qk_kernel<<<Lq * HKC, 128, 0, stream>>>(qb, kb);

    // ---- recurrence + norm + output projection ----------------------------
    gdn_recurrence_kernel<<<dim3(HVC, 2), 256, 0, stream>>>(qb, kb, vb, betab, gb, coreb);
    gated_rmsnorm_kernel<<<Lq * HVC, 128, 0, stream>>>(coreb, zb, norm_w, xnb);
    gemm_bf16_tdm_kernel<<<dim3(2048 / 128, Lq / 256), 256, 0, stream>>>(xnb, WoT, out, Lq, 2048, 4096);
}
